// SelfAttention_68934225101510
// MI455X (gfx1250) — matrix operand order repaired and checked by reading, not compiled, not run on hardware
//
#include <hip/hip_runtime.h>

#define B_ 8
#define S_ 2048
#define H_ 1024

typedef __bf16 v16bf __attribute__((ext_vector_type(16)));
typedef __bf16 v8bf  __attribute__((ext_vector_type(8)));
typedef float  v8f   __attribute__((ext_vector_type(8)));
typedef int    v4i_t __attribute__((vector_size(16)));   // matches async builtin proto

#define GAS __attribute__((address_space(1)))
#define LAS __attribute__((address_space(3)))

// CDNA5 async global->LDS copy path (ASYNCcnt-tracked), guarded for portability.
#if defined(__has_builtin)
#  if __has_builtin(__builtin_amdgcn_global_load_async_to_lds_b128) && \
      __has_builtin(__builtin_amdgcn_s_wait_asynccnt)
#    define USE_ASYNC 1
#  endif
#endif
#ifndef USE_ASYNC
#  define USE_ASYNC 0
#endif

__device__ __forceinline__ unsigned short f32_to_bf16(float f) {
    unsigned int u = __builtin_bit_cast(unsigned int, f);
    u += 0x7FFFu + ((u >> 16) & 1u);   // round-to-nearest-even
    return (unsigned short)(u >> 16);
}

// ---------------- fp32 -> bf16 conversion ----------------
__global__ void __launch_bounds__(256)
cvt_bf16_kernel(const float* __restrict__ in, unsigned short* __restrict__ out, int n) {
    int i = blockIdx.x * 256 + threadIdx.x;
    if (i < n) out[i] = f32_to_bf16(in[i]);
}

// ---------------- WMMA A-fragment loader (direct from global) ----------------
// A frag, contiguous k order used here: lanes 0-15: row=lane,   K = {k0..k0+15}
//                  lanes16-31: row=lane-16,K = {k0+16..k0+31}
__device__ __forceinline__ v16bf load_a_frag(const unsigned short* __restrict__ base,
                                             int lda, int lane, int k0) {
    int row = lane & 15;
    int kb  = k0 + ((lane & 16) ? 16 : 0);
    const unsigned short* p = base + (size_t)row * lda + kb;
    v8bf lo = *(const v8bf*)(p);
    v8bf hi = *(const v8bf*)(p + 8);
    v16bf r;
#pragma unroll
    for (int i = 0; i < 8; ++i) { r[i] = lo[i]; r[i + 8] = hi[i]; }
    return r;
}

// ---------------- Shared GEMM core ----------------
// Wave tile: 32(M) x 64(N), 8 f32 accumulators; block = 8 waves = 256(M) x 64(N).
// B tile (64 cols x 64 K bf16 = 8KB) staged per-block into LDS via CDNA5 async
// copies, double-buffered (16KB LDS). Each settle covers 16 WMMAs.
// LDS tile layout: tile[buf][col*64 + k], col in [0,64), k in [0,64).
__device__ __forceinline__ void gemm_core(const unsigned short* __restrict__ Abase,
                                          const unsigned short* __restrict__ Bbase,
                                          unsigned short (&tile)[2][64 * 64],
                                          int K, int n0, v8f acc[8]) {
    const int lane = threadIdx.x & 31;
    const int t    = threadIdx.x;
    const int scol = t >> 2;       // 0..63  (column this thread stages)
    const int sch  = (t & 3) * 8;  // 0,8,16,24 (first 16B chunk; second at +32)
    const unsigned short* gB = Bbase + (size_t)(n0 + scol) * K + sch;

    auto stage = [&](int buf, int k0) {
#if USE_ASYNC
        __builtin_amdgcn_global_load_async_to_lds_b128(
            (GAS v4i_t*)(gB + k0),
            (LAS v4i_t*)&tile[buf][scol * 64 + sch], 0, 0);
        __builtin_amdgcn_global_load_async_to_lds_b128(
            (GAS v4i_t*)(gB + k0 + 32),
            (LAS v4i_t*)&tile[buf][scol * 64 + sch + 32], 0, 0);
#else
        *(v8bf*)&tile[buf][scol * 64 + sch]      = *(const v8bf*)(gB + k0);
        *(v8bf*)&tile[buf][scol * 64 + sch + 32] = *(const v8bf*)(gB + k0 + 32);
#endif
    };
    auto settle = [&]() {
#if USE_ASYNC
        __builtin_amdgcn_s_wait_asynccnt(0);
#endif
        __syncthreads();
    };

    stage(0, 0);
    settle();
    int buf = 0;
    const int kb = (lane & 16) ? 16 : 0;
    for (int k0 = 0; k0 < K; k0 += 64) {
        if (k0 + 64 < K) stage(buf ^ 1, k0 + 64);
#pragma unroll
        for (int kk = 0; kk < 64; kk += 32) {
            v16bf a0 = load_a_frag(Abase, K, lane, k0 + kk);
            v16bf a1 = load_a_frag(Abase + (size_t)16 * K, K, lane, k0 + kk);
            v16bf b[4];
#pragma unroll
            for (int f = 0; f < 4; ++f)
                b[f] = *(const v16bf*)&tile[buf][(16 * f + (lane & 15)) * 64 + kk + kb];
#pragma unroll
            for (int f = 0; f < 4; ++f) {
                acc[f] = __builtin_amdgcn_wmma_f32_16x16x32_bf16(
                    false, a0, false, b[f], (short)0, acc[f], false, false);
                acc[f + 4] = __builtin_amdgcn_wmma_f32_16x16x32_bf16(
                    false, a1, false, b[f], (short)0, acc[f + 4], false, false);
            }
        }
        if (k0 + 64 < K) settle();   // next tile ready; all waves done with buf
        buf ^= 1;
    }
}

// ---------------- Projection GEMM: Y[m,n] = sum_k X[m,k]*W[n,k] + bias[n] ----------------
// block = 256 thr = 8 waves; wave tile 32(M)x64(N); block tile 256x64
__global__ void __launch_bounds__(256)
proj_gemm_kernel(const unsigned short* __restrict__ A,   // [M,K] bf16
                 const unsigned short* __restrict__ W,   // [N,K] bf16
                 const float* __restrict__ bias,         // [N]
                 unsigned short* __restrict__ out_bf,    // [M,N] or null
                 float* __restrict__ out_f32,            // [M,N] or null
                 int M, int N, int K) {
    __shared__ __align__(32) unsigned short tile[2][64 * 64];
    const int lane = threadIdx.x & 31;
    const int wave = threadIdx.x >> 5;
    const int m0 = blockIdx.x * 256 + wave * 32;
    const int n0 = blockIdx.y * 64;

    v8f acc[8] = {};
    gemm_core(A + (size_t)m0 * K, W, tile, K, n0, acc);

    // epilogue: acc[f+4h] elem e -> row m0+16h+e+(lane>=16?8:0), col n0+16f+(lane&15)
    const int colBase = n0 + (lane & 15);
    const int rowOff  = (lane & 16) ? 8 : 0;
#pragma unroll
    for (int h = 0; h < 2; ++h) {
        const int mbase = m0 + 16 * h + rowOff;
#pragma unroll
        for (int f = 0; f < 4; ++f) {
            const int col = colBase + 16 * f;
            const float bv = bias[col];
#pragma unroll
            for (int e = 0; e < 8; ++e) {
                const size_t idx = (size_t)(mbase + e) * N + col;
                const float val = acc[f + 4 * h][e] + bv;
                if (out_bf) out_bf[idx] = f32_to_bf16(val);
                else        out_f32[idx] = val;
            }
        }
    }
}

// ---------------- Scores: S[b,q,k'] = (1/32)*sum_h Q[b,q,h]*K[b,k',h], masked ----------------
__global__ void __launch_bounds__(256)
scores_kernel(const unsigned short* __restrict__ Q, const unsigned short* __restrict__ Km,
              const unsigned char* __restrict__ mask, float* __restrict__ attn,
              float inv_scale) {
    __shared__ __align__(32) unsigned short tile[2][64 * 64];
    const int b = blockIdx.z;
    const unsigned short* Qb = Q  + (size_t)b * S_ * H_;
    const unsigned short* Kb = Km + (size_t)b * S_ * H_;
    float* Ab = attn + (size_t)b * S_ * S_;
    const unsigned char* mb = mask + (size_t)b * S_;

    const int lane = threadIdx.x & 31;
    const int wave = threadIdx.x >> 5;
    const int q0 = blockIdx.x * 256 + wave * 32;
    const int n0 = blockIdx.y * 64;

    v8f acc[8] = {};
    gemm_core(Qb + (size_t)q0 * H_, Kb, tile, H_, n0, acc);

    const int colBase = n0 + (lane & 15);
    const int rowOff  = (lane & 16) ? 8 : 0;
#pragma unroll
    for (int h = 0; h < 2; ++h) {
        const int qbase = q0 + 16 * h + rowOff;
#pragma unroll
        for (int f = 0; f < 4; ++f) {
            const int col = colBase + 16 * f;
            const bool masked = mb[col] != 0;
#pragma unroll
            for (int e = 0; e < 8; ++e) {
                float val = acc[f + 4 * h][e] * inv_scale;
                if (masked) val = -__builtin_inff();
                Ab[(size_t)(qbase + e) * S_ + col] = val;
            }
        }
    }
}

// ---------------- Row softmax over k (in place), one block per (b,q) row ----------------
__global__ void __launch_bounds__(256)
softmax_kernel(float* __restrict__ attn) {
    __shared__ float red[8];
    float* p = attn + (size_t)blockIdx.x * S_;
    const int tid = threadIdx.x;

    float v[8];
    float m = -__builtin_inff();
#pragma unroll
    for (int j = 0; j < 8; ++j) { v[j] = p[tid + 256 * j]; m = fmaxf(m, v[j]); }
#pragma unroll
    for (int off = 16; off > 0; off >>= 1) m = fmaxf(m, __shfl_down(m, off, 32));
    if ((tid & 31) == 0) red[tid >> 5] = m;
    __syncthreads();
    m = red[0];
#pragma unroll
    for (int w = 1; w < 8; ++w) m = fmaxf(m, red[w]);
    __syncthreads();

    float s = 0.f;
#pragma unroll
    for (int j = 0; j < 8; ++j) { v[j] = __expf(v[j] - m); s += v[j]; }
#pragma unroll
    for (int off = 16; off > 0; off >>= 1) s += __shfl_down(s, off, 32);
    if ((tid & 31) == 0) red[tid >> 5] = s;
    __syncthreads();
    s = red[0];
#pragma unroll
    for (int w = 1; w < 8; ++w) s += red[w];

    const float inv = 1.0f / s;
#pragma unroll
    for (int j = 0; j < 8; ++j) p[tid + 256 * j] = v[j] * inv;
}

// ---------------- Column sums: cs[b,k'] = sum_q attn[b,q,k'] ----------------
__global__ void __launch_bounds__(256)
colsum_kernel(const float* __restrict__ attn, float* __restrict__ cs) {
    const int b = blockIdx.y;
    const int k = blockIdx.x * 256 + threadIdx.x;
    const float* Ab = attn + (size_t)b * S_ * S_;
    float acc = 0.f;
#pragma unroll 4
    for (int q = 0; q < S_; ++q) acc += Ab[(size_t)q * S_ + k];
    cs[b * S_ + k] = acc;
}

// ---------------- context[b,h] = (1/S) * sum_k cs[b,k] * V[b,k,h] ----------------
__global__ void __launch_bounds__(256)
context_kernel(const float* __restrict__ cs, const float* __restrict__ V,
               float* __restrict__ out) {
    const int b = blockIdx.y;
    const int h = blockIdx.x * 256 + threadIdx.x;
    const float* Vb = V + (size_t)b * S_ * H_;
    const float* c  = cs + b * S_;
    float acc = 0.f;
#pragma unroll 4
    for (int k = 0; k < S_; ++k) acc += c[k] * Vb[(size_t)k * H_ + h];
    out[b * H_ + h] = acc * (1.0f / (float)S_);
}

extern "C" void kernel_launch(void* const* d_in, const int* in_sizes, int n_in,
                              void* d_out, int out_size, void* d_ws, size_t ws_size,
                              hipStream_t stream) {
    const float* hidden        = (const float*)d_in[0];
    const unsigned char* mask  = (const unsigned char*)d_in[1];
    const float* Wq = (const float*)d_in[2];
    const float* bq = (const float*)d_in[3];
    const float* Wk = (const float*)d_in[4];
    const float* bk = (const float*)d_in[5];
    const float* Wv = (const float*)d_in[6];
    const float* bv = (const float*)d_in[7];

    float* out_ctx = (float*)d_out;                      // [B,H]
    float* attn    = out_ctx + (size_t)B_ * H_;          // [B,S,S]

    // workspace carve-out
    char* ws = (char*)d_ws;
    size_t off = 0;
    auto alloc = [&](size_t bytes) -> void* {
        void* p = ws + off;
        off += (bytes + 255) & ~(size_t)255;
        return p;
    };
    const size_t BSH = (size_t)B_ * S_ * H_;
    unsigned short* Xbf  = (unsigned short*)alloc(BSH * 2);
    unsigned short* Wqbf = (unsigned short*)alloc((size_t)H_ * H_ * 2);
    unsigned short* Wkbf = (unsigned short*)alloc((size_t)H_ * H_ * 2);
    unsigned short* Wvbf = (unsigned short*)alloc((size_t)H_ * H_ * 2);
    unsigned short* Qbf  = (unsigned short*)alloc(BSH * 2);
    unsigned short* Kbf  = (unsigned short*)alloc(BSH * 2);
    float*          Vf   = (float*)alloc(BSH * 4);
    float*          cs   = (float*)alloc((size_t)B_ * S_ * 4);
    (void)ws_size; (void)in_sizes; (void)n_in; (void)out_size;

    // 1) fp32 -> bf16
    const int nh = B_ * S_ * H_;
    cvt_bf16_kernel<<<(nh + 255) / 256, 256, 0, stream>>>(hidden, Xbf, nh);
    const int nw = H_ * H_;
    cvt_bf16_kernel<<<(nw + 255) / 256, 256, 0, stream>>>(Wq, Wqbf, nw);
    cvt_bf16_kernel<<<(nw + 255) / 256, 256, 0, stream>>>(Wk, Wkbf, nw);
    cvt_bf16_kernel<<<(nw + 255) / 256, 256, 0, stream>>>(Wv, Wvbf, nw);

    // 2) projections (WMMA bf16, fp32 accum). Q,K stored bf16; V stored fp32.
    dim3 gProj((B_ * S_) / 256, H_ / 64);
    proj_gemm_kernel<<<gProj, 256, 0, stream>>>(Xbf, Wqbf, bq, Qbf, nullptr, B_ * S_, H_, H_);
    proj_gemm_kernel<<<gProj, 256, 0, stream>>>(Xbf, Wkbf, bk, Kbf, nullptr, B_ * S_, H_, H_);
    proj_gemm_kernel<<<gProj, 256, 0, stream>>>(Xbf, Wvbf, bv, nullptr, Vf, B_ * S_, H_, H_);

    // 3) scores straight into d_out attn region (scale + mask in epilogue)
    dim3 gSc(S_ / 256, S_ / 64, B_);
    scores_kernel<<<gSc, 256, 0, stream>>>(Qbf, Kbf, mask, attn, 1.0f / 32.0f);

    // 4) softmax rows in place
    softmax_kernel<<<B_ * S_, 256, 0, stream>>>(attn);

    // 5) mean_q(attn @ V) == (colsum_q(attn)/S) @ V
    colsum_kernel<<<dim3(S_ / 256, B_), 256, 0, stream>>>(attn, cs);
    context_kernel<<<dim3(H_ / 256, B_), 256, 0, stream>>>(cs, Vf, out_ctx);
}
